// Fp8Linear_77326591197279
// MI455X (gfx1250) — 8-bit matrix operand order repaired and checked by reading, compiled, not run on hardware
//
#include <hip/hip_runtime.h>
#include <stdint.h>

typedef int   v16i __attribute__((ext_vector_type(16)));
typedef int   v4i  __attribute__((ext_vector_type(4)));
typedef float v8f  __attribute__((ext_vector_type(8)));

union AFrag { v16i v; unsigned long long u[8]; };
union BFrag { v16i v; v4i q[4]; };

#if defined(__has_builtin)
#  if __has_builtin(__builtin_amdgcn_global_load_async_to_lds_b128) && \
      __has_builtin(__builtin_amdgcn_s_wait_asynccnt)
#    define HAS_ASYNC 1
#  endif
#endif
#ifndef HAS_ASYNC
#  define HAS_ASYNC 0
#endif

#if defined(__has_builtin)
#  if __has_builtin(__builtin_amdgcn_sched_barrier)
#    define SCHED_FENCE() __builtin_amdgcn_sched_barrier(0)
#  endif
#endif
#ifndef SCHED_FENCE
#  define SCHED_FENCE()
#endif

#if HAS_ASYNC
typedef __attribute__((address_space(1))) v4i gv4i;
typedef __attribute__((address_space(3))) v4i lv4i;
#endif

__global__ void init_amax_kernel(unsigned* amaxBits) {
    *amaxBits = 0u;
}

__global__ __launch_bounds__(256) void amax_kernel(const float4* __restrict__ x,
                                                   unsigned n4,
                                                   unsigned* __restrict__ amaxBits) {
    float m = 0.0f;
    unsigned stride = gridDim.x * 256u;
    for (unsigned i = blockIdx.x * 256u + threadIdx.x; i < n4; i += stride) {
        float4 v = x[i];
        m = fmaxf(m, fmaxf(fmaxf(fabsf(v.x), fabsf(v.y)),
                           fmaxf(fabsf(v.z), fabsf(v.w))));
    }
#pragma unroll
    for (int off = 16; off > 0; off >>= 1)
        m = fmaxf(m, __shfl_xor(m, off, 32));

    __shared__ float sm[8];
    int lane = threadIdx.x & 31;
    int wave = threadIdx.x >> 5;
    if (lane == 0) sm[wave] = m;
    __syncthreads();
    if (threadIdx.x == 0) {
        float r = sm[0];
#pragma unroll
        for (int i = 1; i < 8; ++i) r = fmaxf(r, sm[i]);
        atomicMax(amaxBits, __float_as_uint(r));
    }
}

__global__ __launch_bounds__(256) void quant_fp8_kernel(const float4* __restrict__ src,
                                                        unsigned* __restrict__ dst,
                                                        unsigned n4,
                                                        const unsigned* __restrict__ amaxBits) {
    float scale = 1.0f;
    if (amaxBits) {
        float amax = fmaxf(__uint_as_float(*amaxBits), 1e-12f);
        scale = 448.0f / amax;
    }
    unsigned stride = gridDim.x * 256u;
    for (unsigned i = blockIdx.x * 256u + threadIdx.x; i < n4; i += stride) {
        float4 v = src[i];
        int p = __builtin_amdgcn_cvt_pk_fp8_f32(v.x * scale, v.y * scale, 0, false);
        p = __builtin_amdgcn_cvt_pk_fp8_f32(v.z * scale, v.w * scale, p, true);
        dst[i] = (unsigned)p;
    }
}

#define BM 128
#define BN 64
#define BK 128
#define LDS_ROW 144
#define SLAB (BN * LDS_ROW)
#define NBUF 3

__device__ __forceinline__ void stage_slab(const uint8_t* __restrict__ gbase,
                                           uint8_t* __restrict__ sdst,
                                           int K, int tid) {
#pragma unroll
    for (int i = 0; i < 2; ++i) {
        const int chunk = tid + i * 256;
        const int row   = chunk >> 3;
        const int kk    = (chunk & 7) * 16;
#if HAS_ASYNC
        __builtin_amdgcn_global_load_async_to_lds_b128(
            (gv4i*)(gbase + (size_t)row * K + kk),
            (lv4i*)(sdst + row * LDS_ROW + kk),
            0, 0);
#else
        *(v4i*)(sdst + row * LDS_ROW + kk) =
            *(const v4i*)(gbase + (size_t)row * K + kk);
#endif
    }
}

__device__ __forceinline__ void load_afrag(AFrag& a, const uint8_t* aRow, int k0) {
#pragma unroll
    for (int j = 0; j < 8; ++j)
        a.u[j] = *(const unsigned long long*)(aRow + k0 + (j >> 1) * 32 + (j & 1) * 8);
}

__global__ __launch_bounds__(256) void gemm_fp8_kernel(const uint8_t* __restrict__ qA,
                                                       const uint8_t* __restrict__ qW,
                                                       const unsigned* __restrict__ amaxBits,
                                                       const float* __restrict__ wScale,
                                                       const float* __restrict__ bias,
                                                       float* __restrict__ out,
                                                       int N, int K) {
    __shared__ uint8_t sB[NBUF * SLAB];

    const int tid  = threadIdx.x;
    const int lane = tid & 31;
    const int wave = tid >> 5;
    const int half = lane >> 4;
    const int l16  = lane & 15;

    const int m0 = blockIdx.x * BM + wave * 16;
    const int n0 = blockIdx.y * BN;

    const uint8_t* aRow  = qA + (size_t)(m0 + l16) * K + half * 16;
    const uint8_t* wBase = qW + (size_t)n0 * K;

    stage_slab(wBase, sB, K, tid);
    if (BK < K) stage_slab(wBase + BK, sB + SLAB, K, tid);

    v8f acc[4] = {};
    int cur = 0;

    AFrag a0, a1;
    load_afrag(a0, aRow, 0);

    auto kstep = [&](AFrag& aCur, AFrag& aNext, int k0) {
#if HAS_ASYNC
        if (k0 + BK < K) __builtin_amdgcn_s_wait_asynccnt(2);
        else             __builtin_amdgcn_s_wait_asynccnt(0);
#endif
        __syncthreads();

        if (k0 + 2 * BK < K) {
            int tgt = cur + 2; if (tgt >= NBUF) tgt -= NBUF;
            stage_slab(wBase + (k0 + 2 * BK), sB + tgt * SLAB, K, tid);
        }

        const uint8_t* sb = sB + cur * SLAB + l16 * LDS_ROW + half * 16;

        BFrag b[4];
#pragma unroll
        for (int t = 0; t < 4; ++t)
#pragma unroll
            for (int c = 0; c < 4; ++c)
                b[t].q[c] = *(const v4i*)(sb + t * 16 * LDS_ROW + c * 32);

        SCHED_FENCE();

        if (k0 + BK < K) load_afrag(aNext, aRow, k0 + BK);

        SCHED_FENCE();

#pragma unroll
        for (int t = 0; t < 4; ++t)
            acc[t] = __builtin_amdgcn_wmma_f32_16x16x128_fp8_fp8(
                aCur.v, b[t].v, (short)0, acc[t], false, false);

        cur = (cur + 1 == NBUF) ? 0 : cur + 1;
    };

    for (int k0 = 0; k0 < K; k0 += 2 * BK) {
        kstep(a0, a1, k0);
        if (k0 + BK < K) kstep(a1, a0, k0 + BK);
    }

    const float amax = fmaxf(__uint_as_float(*amaxBits), 1e-12f);
    const float s = (amax * (1.0f / 448.0f)) * wScale[0];

#pragma unroll
    for (int t = 0; t < 4; ++t) {
        const int n   = n0 + t * 16 + l16;
        const float bv = bias[n];
        float* op = out + (size_t)(m0 + half * 8) * N + n;
#pragma unroll
        for (int r = 0; r < 8; ++r)
            __builtin_nontemporal_store(acc[t][r] * s + bv, op + (size_t)r * N);
    }
}

extern "C" void kernel_launch(void* const* d_in, const int* in_sizes, int n_in,
                              void* d_out, int out_size, void* d_ws, size_t ws_size,
                              hipStream_t stream) {
    const float* input   = (const float*)d_in[0];
    const float* qweight = (const float*)d_in[1];
    const float* wScale  = (const float*)d_in[2];
    const float* bias    = (const float*)d_in[3];

    const int N = in_sizes[3];
    const int K = in_sizes[1] / N;
    const int M = in_sizes[0] / K;

    unsigned* amaxBits = (unsigned*)d_ws;
    uint8_t*  qA = (uint8_t*)d_ws + 256;
    uint8_t*  qW = qA + (size_t)M * K;

    init_amax_kernel<<<1, 1, 0, stream>>>(amaxBits);

    const unsigned n4A = (unsigned)((size_t)M * K / 4);
    const unsigned n4W = (unsigned)((size_t)N * K / 4);

    amax_kernel<<<2048, 256, 0, stream>>>((const float4*)input, n4A, amaxBits);

    quant_fp8_kernel<<<(n4A + 255) / 256, 256, 0, stream>>>(
        (const float4*)input, (unsigned*)qA, n4A, amaxBits);
    quant_fp8_kernel<<<(n4W + 255) / 256, 256, 0, stream>>>(
        (const float4*)qweight, (unsigned*)qW, n4W, nullptr);

    dim3 grid(M / BM, N / BN);
    gemm_fp8_kernel<<<grid, 256, 0, stream>>>(qA, qW, amaxBits, wScale, bias,
                                              (float*)d_out, N, K);
}
